// Classifier_LSTM_26792005993101
// MI455X (gfx1250) — compile-verified, matrix operand order checked by reading, not run on hardware
//
#include <hip/hip_runtime.h>

typedef __attribute__((ext_vector_type(16))) _Float16 v16h;
typedef __attribute__((ext_vector_type(8)))  _Float16 v8h;
typedef __attribute__((ext_vector_type(8)))  float    v8f;
typedef __attribute__((ext_vector_type(4)))  float    v4f;

namespace {
constexpr int kB = 256, kT = 2048, kIn = 27, kH = 200, kCls = 2;
constexpr int kRows    = 16;             // batch rows per block (WMMA M)
constexpr int kKfrag   = 8;              // K = 32(x+bias) + 224(h) = 256 = 8x32
constexpr int kQpad    = 208;            // gate quarter padded 200 -> 208 (52 N-tiles)
constexpr int kWaves   = 16;             // 512 threads -> 4 waves/SIMD -> 256 VGPR budget
constexpr int kThreads = kWaves * 32;
constexpr int kRegTiles = 2;             // tiles w, 16+w register-resident (128 VGPRs)
constexpr int kLdsTiles = 20;            // tiles 32..51 streamed from LDS
constexpr int kAstride = 256 + 8;        // halves per A row (16B pad -> bank spread)
constexpr int kGstride = 20;             // f32 per gate column, col-major [832][20]
constexpr int kABytes  = kRows * kAstride * 2;            // 8448
constexpr int kGBytes  = 4 * kQpad * kGstride * 4;        // 66560
constexpr int kBLBytes = kLdsTiles * kKfrag * 64 * 16;    // 163840
constexpr int kSmemBytes = kABytes + kGBytes + kBLBytes;  // 238848 (< 320 KB WGP LDS)

union AF { v16h v; v8h h2[2]; };

__device__ __forceinline__ float sigf(float x) {
  return 1.0f / (1.0f + __expf(-x));
}
__device__ __forceinline__ float tanhfast(float x) {
  float e = __expf(-2.0f * fabsf(x));
  float r = (1.0f - e) / (1.0f + e);
  return copysignf(r, x);
}

// Bcat = [W_ih pad 27->32 (k=31 is the bias row) ; W_hh pad 200->224].
// Gate quarters padded 200 -> 208; pad columns have zero weights/bias.
__device__ __forceinline__ AF build_bfrag(int tile, int f, int kb, int nloc,
                                          const float* __restrict__ W_ih,
                                          const float* __restrict__ W_hh,
                                          const float* __restrict__ b_ih,
                                          const float* __restrict__ b_hh) {
  const int n = tile * 16 + nloc;
  const int q = n / kQpad;
  const int j = n % kQpad;
  const bool valid = (j < kH);
  const int gr = q * kH + (valid ? j : 0);
  AF b;
  #pragma unroll
  for (int e = 0; e < 16; ++e) {
    const int kk = 32 * f + ((e < 8) ? (kb + e) : (kb + 8 + e));
    float w = 0.0f;
    if (valid) {
      if (kk < kIn) {
        w = W_ih[gr * kIn + kk];
      } else if (kk == 31) {
        w = b_ih[gr] + b_hh[gr];          // bias row: A[m][31] == 1.0
      } else if (kk >= 32) {
        const int kh = kk - 32;
        if (kh < kH) w = W_hh[gr * kH + kh];
      }
    }
    b.h2[e >> 3][e & 7] = (_Float16)w;
  }
  return b;
}
} // namespace

__global__ __launch_bounds__(kThreads, 1) void lstm_fused_kernel(
    const float* __restrict__ inputs,   // [B, T, 27]
    const float* __restrict__ W_ih,     // [800, 27]
    const float* __restrict__ W_hh,     // [800, 200]
    const float* __restrict__ b_ih,     // [800]
    const float* __restrict__ b_hh,     // [800]
    const float* __restrict__ fc_w,     // [2, 200]
    const float* __restrict__ fc_b,     // [2]
    float* __restrict__ out)            // [512 | 256*200 | 256*200] f32
{
  extern __shared__ __align__(16) char smem[];
  _Float16* Abuf = (_Float16*)smem;                       // [16][264] halves: [x|1|h]
  float*    Gbuf = (float*)(smem + kABytes);              // [832][20] f32 col-major
  _Float16* BL   = (_Float16*)(smem + kABytes + kGBytes); // 20 streamed B tiles

  const int tid  = threadIdx.x;
  const int lane = tid & 31;
  const int wave = tid >> 5;
  const int row0 = blockIdx.x * kRows;

  const int nloc  = lane & 15;
  const int kb    = (lane >> 4) << 3;   // A/B fragment K-half base
  const int mrowC = (lane >> 4) << 3;   // C fragment row base

  // ---- register-resident B fragments: tiles w and 16+w ----
  AF bfr[kRegTiles][kKfrag];
  #pragma unroll
  for (int tt = 0; tt < kRegTiles; ++tt) {
    const int tile = wave + 16 * tt;
    #pragma unroll
    for (int f = 0; f < kKfrag; ++f)
      bfr[tt][f] = build_bfrag(tile, f, kb, nloc, W_ih, W_hh, b_ih, b_hh);
  }
  // ---- LDS-resident B tiles 32..51 (ti = tile - 32) ----
  {
    const int tile = 32 + wave;         // ti = wave
    #pragma unroll
    for (int f = 0; f < kKfrag; ++f) {
      AF b = build_bfrag(tile, f, kb, nloc, W_ih, W_hh, b_ih, b_hh);
      const int base = ((wave * kKfrag + f) * 64 + lane) * 8;
      *(v8h*)&BL[base]       = b.h2[0];
      *(v8h*)&BL[base + 256] = b.h2[1];
    }
  }
  if (wave < 4) {
    const int tile = 48 + wave;         // ti = 16 + wave
    #pragma unroll
    for (int f = 0; f < kKfrag; ++f) {
      AF b = build_bfrag(tile, f, kb, nloc, W_ih, W_hh, b_ih, b_hh);
      const int base = (((16 + wave) * kKfrag + f) * 64 + lane) * 8;
      *(v8h*)&BL[base]       = b.h2[0];
      *(v8h*)&BL[base + 256] = b.h2[1];
    }
  }

  // ---- zero A (h0 = 0; pad columns stay 0), bias column, cell state ----
  for (int i = tid; i < kRows * kAstride; i += kThreads) Abuf[i] = (_Float16)0.0f;
  float cS[7];
  #pragma unroll
  for (int k = 0; k < 7; ++k) cS[k] = 0.0f;
  __syncthreads();

  if (tid < kRows * kIn) {  // stage x_0
    const int m = tid / kIn, k = tid % kIn;
    Abuf[m * kAstride + k] = (_Float16)inputs[(size_t)(row0 + m) * kT * kIn + k];
  }
  if (tid < kRows) Abuf[tid * kAstride + 31] = (_Float16)1.0f;  // bias row
  __syncthreads();

  const int mrowU = tid & 15;   // update-phase row
  const int jb    = tid >> 4;   // update-phase j base (0..31)

  for (int t = 0; t < kT; ++t) {
    // --- gates = [x_t | 1 | h_{t-1}] @ Bcat : 2 reg tiles + 1 LDS tile inline ---
    v8f acc[3] = {};
    #pragma unroll
    for (int f = 0; f < kKfrag; ++f) {
      AF a, bl;
      const int abase = (lane & 15) * kAstride + 32 * f + kb;
      a.h2[0] = *(const v8h*)&Abuf[abase];
      a.h2[1] = *(const v8h*)&Abuf[abase + 16];
      const int bbase = ((wave * kKfrag + f) * 64 + lane) * 8;
      bl.h2[0] = *(const v8h*)&BL[bbase];
      bl.h2[1] = *(const v8h*)&BL[bbase + 256];
      acc[0] = __builtin_amdgcn_wmma_f32_16x16x32_f16(
          false, a.v, false, bfr[0][f].v, (short)0, acc[0], false, false);
      acc[1] = __builtin_amdgcn_wmma_f32_16x16x32_f16(
          false, a.v, false, bfr[1][f].v, (short)0, acc[1], false, false);
      acc[2] = __builtin_amdgcn_wmma_f32_16x16x32_f16(
          false, a.v, false, bl.v, (short)0, acc[2], false, false);
    }
    #pragma unroll
    for (int tt = 0; tt < 3; ++tt) {
      const int n = (wave + 16 * tt) * 16 + nloc;   // tiles w, 16+w, 32+w
      float* g = &Gbuf[n * kGstride + mrowC];
      v4f lo = {acc[tt][0], acc[tt][1], acc[tt][2], acc[tt][3]};
      v4f hi = {acc[tt][4], acc[tt][5], acc[tt][6], acc[tt][7]};
      *(v4f*)&g[0] = lo;
      *(v4f*)&g[4] = hi;
    }
    // --- tiles 48..51: second LDS-streamed pass (waves 0..3 only) ---
    if (wave < 4) {
      v8f a3 = {};
      #pragma unroll
      for (int f = 0; f < kKfrag; ++f) {
        AF a, bl;
        const int abase = (lane & 15) * kAstride + 32 * f + kb;
        a.h2[0] = *(const v8h*)&Abuf[abase];
        a.h2[1] = *(const v8h*)&Abuf[abase + 16];
        const int bbase = (((16 + wave) * kKfrag + f) * 64 + lane) * 8;
        bl.h2[0] = *(const v8h*)&BL[bbase];
        bl.h2[1] = *(const v8h*)&BL[bbase + 256];
        a3 = __builtin_amdgcn_wmma_f32_16x16x32_f16(
            false, a.v, false, bl.v, (short)0, a3, false, false);
      }
      const int n = (48 + wave) * 16 + nloc;
      float* g = &Gbuf[n * kGstride + mrowC];
      v4f lo = {a3[0], a3[1], a3[2], a3[3]};
      v4f hi = {a3[4], a3[5], a3[6], a3[7]};
      *(v4f*)&g[0] = lo;
      *(v4f*)&g[4] = hi;
    }
    __syncthreads();   // gates visible; all A reads complete

    // --- LSTM cell update: thread -> (m = tid&15, j = jb + 32k) ---
    #pragma unroll
    for (int k = 0; k < 7; ++k) {
      const int j = jb + 32 * k;
      if (j < kQpad) {
        const float i_ = sigf(Gbuf[(0 * kQpad + j) * kGstride + mrowU]);
        const float f_ = sigf(Gbuf[(1 * kQpad + j) * kGstride + mrowU]);
        const float g_ = tanhfast(Gbuf[(2 * kQpad + j) * kGstride + mrowU]);
        const float o_ = sigf(Gbuf[(3 * kQpad + j) * kGstride + mrowU]);
        const float c_ = f_ * cS[k] + i_ * g_;
        cS[k] = c_;
        if (j < kH) Abuf[mrowU * kAstride + 32 + j] = (_Float16)(o_ * tanhfast(c_));
      }
    }

    // --- stage x_{t+1}; prefetch x_{t+2} stream ---
    if (t + 1 < kT) {
      if (tid < kRows * kIn) {
        const int m = tid / kIn, k = tid % kIn;
        Abuf[m * kAstride + k] = (_Float16)
            inputs[(size_t)(row0 + m) * kT * kIn + (size_t)(t + 1) * kIn + k];
      }
      if (tid < kRows && t + 2 < kT) {
        __builtin_prefetch(
            &inputs[(size_t)(row0 + tid) * kT * kIn + (size_t)(t + 2) * kIn], 0, 0);
      }
    }
    __syncthreads();   // A ready for next step; gate reads complete
  }

  // ---- outputs: out[B,2] | h_last[1,B,200] | c_last[1,B,200] ----
  float* out_h = out + kB * kCls;
  float* out_c = out_h + kB * kH;
  #pragma unroll
  for (int k = 0; k < 7; ++k) {
    const int j = jb + 32 * k;
    if (j < kH) {
      out_h[(size_t)(row0 + mrowU) * kH + j] = (float)Abuf[mrowU * kAstride + 32 + j];
      out_c[(size_t)(row0 + mrowU) * kH + j] = cS[k];
    }
  }
  if (wave == 0) {  // FC head: 16 rows x 2 classes on 32 lanes
    const int m = lane & 15, cls = lane >> 4;
    float acc = fc_b[cls];
    for (int j = 0; j < kH; ++j)
      acc += (float)Abuf[m * kAstride + 32 + j] * fc_w[cls * kH + j];
    out[(row0 + m) * kCls + cls] = sigf(acc);
  }
}

extern "C" void kernel_launch(void* const* d_in, const int* in_sizes, int n_in,
                              void* d_out, int out_size, void* d_ws, size_t ws_size,
                              hipStream_t stream) {
  (void)in_sizes; (void)n_in; (void)out_size; (void)d_ws; (void)ws_size;
  const float* inputs = (const float*)d_in[0];
  // d_in[1] X_lengths, d_in[2] h0, d_in[3] c0: unused by the reference math
  const float* W_ih = (const float*)d_in[4];
  const float* W_hh = (const float*)d_in[5];
  const float* b_ih = (const float*)d_in[6];
  const float* b_hh = (const float*)d_in[7];
  const float* fc_w = (const float*)d_in[8];
  const float* fc_b = (const float*)d_in[9];
  lstm_fused_kernel<<<dim3(kB / kRows), dim3(kThreads), kSmemBytes, stream>>>(
      inputs, W_ih, W_hh, b_ih, b_hh, fc_w, fc_b, (float*)d_out);
}
